// DWT_MLP_Model_79104707658288
// MI455X (gfx1250) — compile-verified, matrix operand order checked by reading, not run on hardware
//
#include <hip/hip_runtime.h>
#include <hip/hip_bf16.h>

typedef __attribute__((ext_vector_type(16))) _Float16 v16h;
typedef __attribute__((ext_vector_type(8)))  _Float16 v8h;
typedef __attribute__((ext_vector_type(8)))  float    v8f;

#define BS      4
#define SEQ     2048
#define DMODEL  512
#define NHEAD   8
#define DK      64
#define WINDOW  1024

// ---------------------------------------------------------------------------
// f32 -> f16 elementwise convert
// ---------------------------------------------------------------------------
__global__ void __launch_bounds__(256)
conv_f16_kernel(const float* __restrict__ in, _Float16* __restrict__ out, int n) {
    int i = blockIdx.x * 256 + threadIdx.x;
    if (i < n) out[i] = (_Float16)in[i];
}

// ---------------------------------------------------------------------------
// W[k][n] (512x512 f32) -> Wt[n][k] (f16): makes B-fragment loads contiguous
// ---------------------------------------------------------------------------
__global__ void __launch_bounds__(256)
transpose_f16_kernel(const float* __restrict__ w, _Float16* __restrict__ wt) {
    int i = blockIdx.x * 256 + threadIdx.x;      // i = n*512 + k
    int n = i >> 9;
    int k = i & 511;
    wt[i] = (_Float16)w[k * DMODEL + n];
}

// ---------------------------------------------------------------------------
// WMMA GEMM: C[M,512] = A[M,512](f16) @ Wt[n][k](f16)^T + bias[n]
// One 16x16 tile per wave, K-loop step 32 (v_wmma_f32_16x16x32_f16).
// A-fragment lane pattern (16-bit A 16x32, ISA 7.12.2):
//   lanes 0-15: row=lane,   K = {0..7, 16..23}
//   lanes 16-31: row=lane-16, K = {8..15, 24..31}
// B-fragment mirrors it with lane -> column n, sourced from Wt rows.
// MODE 0: f16 out [b,h,S,dk]   (Q, K)
// MODE 1: f16 out [b,h,dk,S]   (V transposed, for P@V B-fragments)
// MODE 2: f32 out [M,512]      (final projection -> d_out)
// ---------------------------------------------------------------------------
template<int MODE>
__global__ void __launch_bounds__(256)
gemm_wmma_kernel(const _Float16* __restrict__ A,
                 const _Float16* __restrict__ Wt,
                 const float*    __restrict__ bias,
                 void*           __restrict__ out,
                 int M) {
    const int lane = threadIdx.x & 31;
    const int wave = threadIdx.x >> 5;
    const int half = lane >> 4;   // which K sub-pattern
    const int r    = lane & 15;   // A: row-in-tile, B: col-in-tile

    const int tilesN = DMODEL / 16;                 // 32
    int tile = blockIdx.x * 8 + wave;
    int tm = tile / tilesN;
    int tn = tile % tilesN;
    if (tm * 16 >= M) return;
    const int m0 = tm * 16, n0 = tn * 16;

    v8f acc;
    {
        float bv = bias[n0 + r];                    // C layout: col n = n0 + (lane&15)
        #pragma unroll
        for (int i = 0; i < 8; ++i) acc[i] = bv;
    }

    const _Float16* arow = A  + (size_t)(m0 + r) * DMODEL;
    const _Float16* brow = Wt + (size_t)(n0 + r) * DMODEL;

    for (int kb = 0; kb < DMODEL; kb += 32) {
        v8h alo = *(const v8h*)(arow + kb + half * 8);
        v8h ahi = *(const v8h*)(arow + kb + 16 + half * 8);
        v8h blo = *(const v8h*)(brow + kb + half * 8);
        v8h bhi = *(const v8h*)(brow + kb + 16 + half * 8);
        v16h a, b;
        #pragma unroll
        for (int i = 0; i < 8; ++i) {
            a[i] = alo[i]; a[8 + i] = ahi[i];
            b[i] = blo[i]; b[8 + i] = bhi[i];
        }
        acc = __builtin_amdgcn_wmma_f32_16x16x32_f16(
            false, a, false, b, (short)0, acc, false, false);
    }

    #pragma unroll
    for (int p = 0; p < 8; ++p) {
        int m = m0 + p + half * 8;   // C layout: VGPR p -> row p (lanes 0-15) / p+8
        int n = n0 + r;
        float v = acc[p];
        if (MODE == 0) {
            int b_ = m >> 11, s = m & 2047, h = n >> 6, d = n & 63;
            ((_Float16*)out)[(((size_t)(b_ * NHEAD + h) * SEQ + s) * DK) + d] = (_Float16)v;
        } else if (MODE == 1) {
            int b_ = m >> 11, s = m & 2047, h = n >> 6, d = n & 63;
            ((_Float16*)out)[(((size_t)(b_ * NHEAD + h) * DK + d) * SEQ) + s] = (_Float16)v;
        } else {
            ((float*)out)[(size_t)m * DMODEL + n] = v;
        }
    }
}

// ---------------------------------------------------------------------------
// Banded causal flash attention: mask is 0 <= i-j <= 1024 (log-sparse mask
// degenerates to the neighbor band for k = S/2). One wave = one 16-query
// tile of one (b,h). j advances 32 per step: 4 QK^T WMMAs + 4 P@V WMMAs.
// P transposed C-layout -> A-fragment through per-wave LDS.
// ---------------------------------------------------------------------------
__global__ void __launch_bounds__(256)
attn_wmma_kernel(const _Float16* __restrict__ Q,    // [b,h,S,dk]
                 const _Float16* __restrict__ K,    // [b,h,S,dk]
                 const _Float16* __restrict__ Vt,   // [b,h,dk,S]
                 _Float16*       __restrict__ out)  // [b,S,D] f16
{
    __shared__ _Float16 lds[8][16][32];             // per-wave P staging (8 KB)

    const int lane = threadIdx.x & 31;
    const int wave = threadIdx.x >> 5;
    const int half = lane >> 4;
    const int r    = lane & 15;

    const int tilesPerHead = SEQ / 16;              // 128
    int tile = blockIdx.x * 8 + wave;
    int bh = tile / tilesPerHead;                   // 0..31
    int qt = tile % tilesPerHead;
    int b  = bh / NHEAD, h = bh % NHEAD;
    const int i0 = qt * 16;

    const _Float16* Qbh = Q  + (size_t)bh * SEQ * DK;
    const _Float16* Kbh = K  + (size_t)bh * SEQ * DK;
    const _Float16* Vbh = Vt + (size_t)bh * DK * SEQ;

    // Q fragments for d in [0,32) and [32,64), pre-scaled by 1/sqrt(dk)=0.125
    v16h qa0, qa1;
    {
        const _Float16* qrow = Qbh + (size_t)(i0 + r) * DK;
        v8h q0 = *(const v8h*)(qrow + half * 8);
        v8h q1 = *(const v8h*)(qrow + 16 + half * 8);
        v8h q2 = *(const v8h*)(qrow + 32 + half * 8);
        v8h q3 = *(const v8h*)(qrow + 48 + half * 8);
        const _Float16 sc = (_Float16)0.125f;
        #pragma unroll
        for (int i = 0; i < 8; ++i) {
            qa0[i] = q0[i] * sc;  qa0[8 + i] = q1[i] * sc;
            qa1[i] = q2[i] * sc;  qa1[8 + i] = q3[i] * sc;
        }
    }

    float mrow[8], lrow[8];
    v8f o[4];
    #pragma unroll
    for (int p = 0; p < 8; ++p) { mrow[p] = -INFINITY; lrow[p] = 0.0f; }
    #pragma unroll
    for (int c = 0; c < 4; ++c)
        #pragma unroll
        for (int p = 0; p < 8; ++p) o[c][p] = 0.0f;

    int jstart = i0 - WINDOW; if (jstart < 0) jstart = 0;
    jstart &= ~31;

    for (int j0 = jstart; j0 <= i0 + 15; j0 += 32) {
        // ---- scores: two 16x16 tiles (cols j0..j0+15 and j0+16..j0+31) ----
        v8f s0, s1;
        #pragma unroll
        for (int p = 0; p < 8; ++p) { s0[p] = 0.0f; s1[p] = 0.0f; }
        {
            const _Float16* krow = Kbh + (size_t)(j0 + r) * DK;
            v8h k0 = *(const v8h*)(krow + half * 8);
            v8h k1 = *(const v8h*)(krow + 16 + half * 8);
            v8h k2 = *(const v8h*)(krow + 32 + half * 8);
            v8h k3 = *(const v8h*)(krow + 48 + half * 8);
            v16h kb0, kb1;
            #pragma unroll
            for (int i = 0; i < 8; ++i) {
                kb0[i] = k0[i]; kb0[8 + i] = k1[i];
                kb1[i] = k2[i]; kb1[8 + i] = k3[i];
            }
            s0 = __builtin_amdgcn_wmma_f32_16x16x32_f16(false, qa0, false, kb0, (short)0, s0, false, false);
            s0 = __builtin_amdgcn_wmma_f32_16x16x32_f16(false, qa1, false, kb1, (short)0, s0, false, false);
        }
        {
            const _Float16* krow = Kbh + (size_t)(j0 + 16 + r) * DK;
            v8h k0 = *(const v8h*)(krow + half * 8);
            v8h k1 = *(const v8h*)(krow + 16 + half * 8);
            v8h k2 = *(const v8h*)(krow + 32 + half * 8);
            v8h k3 = *(const v8h*)(krow + 48 + half * 8);
            v16h kb0, kb1;
            #pragma unroll
            for (int i = 0; i < 8; ++i) {
                kb0[i] = k0[i]; kb0[8 + i] = k1[i];
                kb1[i] = k2[i]; kb1[8 + i] = k3[i];
            }
            s1 = __builtin_amdgcn_wmma_f32_16x16x32_f16(false, qa0, false, kb0, (short)0, s1, false, false);
            s1 = __builtin_amdgcn_wmma_f32_16x16x32_f16(false, qa1, false, kb1, (short)0, s1, false, false);
        }

        // ---- band mask (skip for interior tiles) ----
        bool needMask = (j0 + 31 > i0) || (j0 < i0 + 15 - WINDOW);
        if (needMask) {
            #pragma unroll
            for (int p = 0; p < 8; ++p) {
                int i = i0 + p + half * 8;
                int d0 = i - (j0 + r);
                int d1 = i - (j0 + 16 + r);
                if (d0 < 0 || d0 > WINDOW) s0[p] = -INFINITY;
                if (d1 < 0 || d1 > WINDOW) s1[p] = -INFINITY;
            }
        }

        // ---- online softmax (row reductions over 16-lane groups) ----
        #pragma unroll
        for (int p = 0; p < 8; ++p) {
            float t = fmaxf(s0[p], s1[p]);
            #pragma unroll
            for (int off = 1; off < 16; off <<= 1)
                t = fmaxf(t, __shfl_xor(t, off, 32));
            float mn = fmaxf(fmaxf(mrow[p], t), -1e30f);  // clamp: fully-masked tile safe
            float scale = __expf(mrow[p] - mn);
            mrow[p] = mn;
            float p0 = __expf(s0[p] - mn);
            float p1 = __expf(s1[p] - mn);
            float rs = p0 + p1;
            #pragma unroll
            for (int off = 1; off < 16; off <<= 1)
                rs += __shfl_xor(rs, off, 32);
            lrow[p] = lrow[p] * scale + rs;
            #pragma unroll
            for (int c = 0; c < 4; ++c) o[c][p] *= scale;
            // stage P in C layout (row = p + half*8, cols = r and r+16)
            lds[wave][p + half * 8][r]      = (_Float16)p0;
            lds[wave][p + half * 8][16 + r] = (_Float16)p1;
        }

        // same-wave LDS ordering: make staged P visible before re-load
        asm volatile("s_wait_dscnt 0x0" ::: "memory");

        // ---- reload P as 16x32 A-fragment ----
        v16h pa;
        {
            const _Float16* prow = &lds[wave][r][0];
            v8h plo = *(const v8h*)(prow + half * 8);
            v8h phi = *(const v8h*)(prow + 16 + half * 8);
            #pragma unroll
            for (int i = 0; i < 8; ++i) { pa[i] = plo[i]; pa[8 + i] = phi[i]; }
        }

        // ---- O += P @ V (4 dk-chunks of 16, B-fragments from Vt rows) ----
        #pragma unroll
        for (int c = 0; c < 4; ++c) {
            const _Float16* vrow = Vbh + (size_t)(c * 16 + r) * SEQ + j0;
            v8h v0 = *(const v8h*)(vrow + half * 8);
            v8h v1 = *(const v8h*)(vrow + 16 + half * 8);
            v16h vb;
            #pragma unroll
            for (int i = 0; i < 8; ++i) { vb[i] = v0[i]; vb[8 + i] = v1[i]; }
            o[c] = __builtin_amdgcn_wmma_f32_16x16x32_f16(
                false, pa, false, vb, (short)0, o[c], false, false);
        }
    }

    // ---- finalize: divide by l, store f16 [b,S,D] ----
    #pragma unroll
    for (int p = 0; p < 8; ++p) {
        float inv = 1.0f / lrow[p];
        int srow = i0 + p + half * 8;
        _Float16* orow = out + ((size_t)b * SEQ + srow) * DMODEL + h * DK;
        #pragma unroll
        for (int c = 0; c < 4; ++c)
            orow[c * 16 + r] = (_Float16)(o[c][p] * inv);
    }
}

// ---------------------------------------------------------------------------
// Launch
// ---------------------------------------------------------------------------
extern "C" void kernel_launch(void* const* d_in, const int* in_sizes, int n_in,
                              void* d_out, int out_size, void* d_ws, size_t ws_size,
                              hipStream_t stream) {
    const float* x  = (const float*)d_in[0];
    const float* WQ = (const float*)d_in[1];
    const float* bQ = (const float*)d_in[2];
    const float* WK = (const float*)d_in[3];
    const float* bK = (const float*)d_in[4];
    const float* WV = (const float*)d_in[5];
    const float* bV = (const float*)d_in[6];
    const float* WO = (const float*)d_in[7];
    const float* bO = (const float*)d_in[8];

    const int M = BS * SEQ;                         // 8192
    char* ws = (char*)d_ws;
    size_t off = 0;
    _Float16* xh  = (_Float16*)(ws + off); off += (size_t)M * DMODEL * 2;          // 8 MiB
    _Float16* wqt = (_Float16*)(ws + off); off += (size_t)DMODEL * DMODEL * 2;
    _Float16* wkt = (_Float16*)(ws + off); off += (size_t)DMODEL * DMODEL * 2;
    _Float16* wvt = (_Float16*)(ws + off); off += (size_t)DMODEL * DMODEL * 2;
    _Float16* wot = (_Float16*)(ws + off); off += (size_t)DMODEL * DMODEL * 2;
    _Float16* Qh  = (_Float16*)(ws + off); off += (size_t)M * DMODEL * 2;
    _Float16* Kh  = (_Float16*)(ws + off); off += (size_t)M * DMODEL * 2;
    _Float16* Vth = (_Float16*)(ws + off); off += (size_t)M * DMODEL * 2;
    _Float16* Ah  = (_Float16*)(ws + off); off += (size_t)M * DMODEL * 2;          // attn out

    // 1) convert x, transpose+convert weights
    conv_f16_kernel<<<(M * DMODEL + 255) / 256, 256, 0, stream>>>(x, xh, M * DMODEL);
    transpose_f16_kernel<<<(DMODEL * DMODEL) / 256, 256, 0, stream>>>(WQ, wqt);
    transpose_f16_kernel<<<(DMODEL * DMODEL) / 256, 256, 0, stream>>>(WK, wkt);
    transpose_f16_kernel<<<(DMODEL * DMODEL) / 256, 256, 0, stream>>>(WV, wvt);
    transpose_f16_kernel<<<(DMODEL * DMODEL) / 256, 256, 0, stream>>>(WO, wot);

    // 2) projections
    const int gemmBlocks = (M / 16) * (DMODEL / 16) / 8;                           // 2048
    gemm_wmma_kernel<0><<<gemmBlocks, 256, 0, stream>>>(xh, wqt, bQ, Qh, M);
    gemm_wmma_kernel<0><<<gemmBlocks, 256, 0, stream>>>(xh, wkt, bK, Kh, M);
    gemm_wmma_kernel<1><<<gemmBlocks, 256, 0, stream>>>(xh, wvt, bV, Vth, M);

    // 3) banded flash attention (BS*NHEAD*SEQ/16 = 4096 tiles, 8 waves/block)
    attn_wmma_kernel<<<4096 / 8, 256, 0, stream>>>(Qh, Kh, Vth, Ah);

    // 4) output projection -> f32 d_out
    gemm_wmma_kernel<2><<<gemmBlocks, 256, 0, stream>>>(Ah, wot, bO, d_out, M);
}
